// LinearGRPOLoss_20916490731693
// MI455X (gfx1250) — matrix operand order repaired and checked by reading, not compiled, not run on hardware
//
#include <hip/hip_runtime.h>

// ---------------------------------------------------------------------------
// Fused linear + GRPO loss for gfx1250 (MI455X).
//
// Compute bound (≈3200 FLOP/B): bf16 WMMA 16x16x32 with hi/lo split (3 WMMAs)
// for ~fp32 accuracy at matrix-core rate. Lane-local online logsumexp +
// predicated target accumulate -> no cross-lane ops / barriers in the sweep;
// the [8192 x 32000] logit matrix never touches memory.
//
// Fast path: W split/swizzled once per launch into per-lane WMMA B-fragment
// order. One 256-thread block owns 32 tokens (A slab ~263KB LDS, 1 block/WGP)
// as a 2x4 wave grid; wave-rows share B fragments through WGP$/L2, halving
// per-WGP weight traffic vs two 16-token blocks. Per K-stage per wave:
// 4x ds_load_b128 + 8x global_load_b128 + 6x v_wmma.
// ---------------------------------------------------------------------------

typedef __attribute__((ext_vector_type(16))) __bf16 v16bf;
typedef __attribute__((ext_vector_type(8)))  __bf16 bf8v;
typedef __attribute__((ext_vector_type(8)))  float  v8f;
typedef __attribute__((ext_vector_type(8)))  unsigned short us8v;

#define THREADS_F 256          // fast path: 8 waves, 2 rows x 4 cols
#define M_FAST    32           // tokens per fast block
#define THREADS_S 128          // slow path: 4 waves
#define M_SLOW    16
#define NB_FAST   128          // vocab cols per sweep iter (4 cols x 2 tiles x 16)
#define NB_SLOW   64
#define KC        32           // K per WMMA stage
#define EMB_C     2048
#define NKC_C     (EMB_C / KC) // 64 stages
#define A_STRIDE  (EMB_C + 8)  // LDS row padding -> conflict-free ds_load_b128
#define B_STRIDE  (KC + 8)
#define FRAG_ELEMS 512         // 16 cols x 32 k bf16 per fragment (1KB)
#define KLC       0.1f

__device__ __forceinline__ void split_bf(float x, unsigned short& h, unsigned short& l) {
  unsigned u = __float_as_uint(x);
  h = (unsigned short)(u >> 16);                       // truncated bf16 high part
  float r = x - __uint_as_float(u & 0xffff0000u);      // exact residual
  l = (unsigned short)((__float_as_uint(r) + 0x8000u) >> 16);  // rounded bf16 low part
}

// Two 16B loads (LDS or global, address-space generic) -> one v16bf fragment.
__device__ __forceinline__ v16bf mk16(const unsigned short* p0, const unsigned short* p1) {
  bf8v a = *(const bf8v*)p0;
  bf8v b = *(const bf8v*)p1;
  v16bf f;
#pragma unroll
  for (int i = 0; i < 8; ++i) { f[i] = a[i]; f[i + 8] = b[i]; }
  return f;
}

// ---------------------------------------------------------------------------
// Prep: split W[VOCAB][EMB] f32 into bf16 hi/lo, swizzled fragment-major:
// fragment(nt16, s) = B frag for cols [nt16*16,+16), k [s*32,+32);
// lane l owns elements [l*16, l*16+16): n = nt16*16+(l&15),
// k = s*32 + ((l&16)?16:0) + 0..15  (contiguous k order, same as the A reads).
// ---------------------------------------------------------------------------
__global__ __launch_bounds__(256)
void grpo_split_w(const float* __restrict__ W, unsigned short* __restrict__ Whi,
                  unsigned short* __restrict__ Wlo, int nfrag) {
  int frag = blockIdx.x * (blockDim.x >> 5) + (threadIdx.x >> 5);
  if (frag >= nfrag) return;
  int lane = threadIdx.x & 31;
  int nt16 = frag / NKC_C;
  int s    = frag - nt16 * NKC_C;
  int n    = nt16 * 16 + (lane & 15);
  int k0   = s * KC + ((lane & 16) ? 16 : 0);
  const float* src = W + (size_t)n * EMB_C + k0;   // 64B aligned
  unsigned short h[16], l[16];
#pragma unroll
  for (int j = 0; j < 16; ++j) split_bf(src[j], h[j], l[j]);
  size_t off = (size_t)frag * FRAG_ELEMS + (size_t)lane * 16;
  *(us8v*)(Whi + off)     = *(const us8v*)&h[0];
  *(us8v*)(Whi + off + 8) = *(const us8v*)&h[8];
  *(us8v*)(Wlo + off)     = *(const us8v*)&l[0];
  *(us8v*)(Wlo + off + 8) = *(const us8v*)&l[8];
}

// ---------------------------------------------------------------------------
// Shared helpers
// ---------------------------------------------------------------------------

// Lane-local online-LSE fold of one 16x16 logit tile (C layout: vgpr r ->
// M = r (lanes 0-15) / 8+r (lanes 16-31), N = lane&15). No cross-lane ops.
__device__ __forceinline__ void fold_tile(const v8f& acc, int n_g, const int* tgtIdx,
                                          float* runMx, float* runSm, float* tgtAcc) {
#pragma unroll
  for (int r = 0; r < 8; ++r) {
    float v = acc[r];
    tgtAcc[r] += (tgtIdx[r] == n_g) ? v : 0.0f;      // one lane ever matches
    float nm = fmaxf(runMx[r], v);
    runSm[r] = runSm[r] * __expf(runMx[r] - nm) + __expf(v - nm);
    runMx[r] = nm;
  }
}

// One-time 16-lane butterfly reduction of the per-lane state.
__device__ __forceinline__ void lane_reduce(float* runMx, float* runSm, float* tgtAcc) {
#pragma unroll
  for (int r = 0; r < 8; ++r) {
    float mx = runMx[r];
#pragma unroll
    for (int d = 1; d < 16; d <<= 1) mx = fmaxf(mx, __shfl_xor(mx, d));
    float sm = runSm[r] * __expf(runMx[r] - mx);
    float tg = tgtAcc[r];
#pragma unroll
    for (int d = 1; d < 16; d <<= 1) { sm += __shfl_xor(sm, d); tg += __shfl_xor(tg, d); }
    runMx[r] = mx; runSm[r] = sm; tgtAcc[r] = tg;
  }
}

__device__ __forceinline__ void token_out(float M, float S, float T, int tok,
                                          const float* ref_lp, const float* adv,
                                          const unsigned char* pmask,
                                          float* out, float* ws, int SEQ) {
  float logZ = M + logf(S);
  float pi   = T - logZ;
  out[5 + tok] = pi;                               // pi_logprobs

  int   b     = tok / SEQ;
  float mask  = (pmask[tok] != 0) ? 1.0f : 0.0f;   // bool input, 1 byte/elem
  float a     = adv[b];
  float delta = ref_lp[tok] - pi;
  float kl    = expf(delta) - delta - 1.0f;
  float pol   = a;                                 // exp(pi - sg(pi)) == 1 forward
  float lss   = -(pol - KLC * kl);
  atomicAdd(&ws[b * 4 + 0], lss * mask);
  atomicAdd(&ws[b * 4 + 1], kl  * mask);
  atomicAdd(&ws[b * 4 + 2], pol * mask);
  atomicAdd(&ws[b * 4 + 3], mask);
}

// ---------------------------------------------------------------------------
// FAST PATH: 256 threads, 32 tokens, 2x4 wave grid, two 16-col tiles/wave.
// ---------------------------------------------------------------------------
__global__ __launch_bounds__(THREADS_F)
void grpo_fused_fast(const unsigned short* __restrict__ Whi,
                     const unsigned short* __restrict__ Wlo,
                     const float* __restrict__ H,
                     const int* __restrict__ targets, const float* __restrict__ ref_lp,
                     const float* __restrict__ adv, const unsigned char* __restrict__ pmask,
                     float* __restrict__ out, float* __restrict__ ws,
                     int VOCAB, int SEQ)
{
  __shared__ __align__(16) unsigned short Ahi[M_FAST][A_STRIDE];   // 128.5 KB
  __shared__ __align__(16) unsigned short Alo[M_FAST][A_STRIDE];   // 128.5 KB
  __shared__ float pMx[4][M_FAST];
  __shared__ float pSm[4][M_FAST];
  __shared__ float pTg[4][M_FAST];

  const int tid    = threadIdx.x;
  const int wave   = tid >> 5;
  const int lane   = tid & 31;
  const int wr     = wave >> 2;        // wave row: tokens [wr*16, +16)
  const int wc     = wave & 3;         // wave col
  const int m_base = blockIdx.x * M_FAST;

  // ---- Stage the 32-token activation slab once, split into bf16 hi/lo ----
  {
    const float4* src = (const float4*)(H + (size_t)m_base * EMB_C);
#pragma unroll 4
    for (int i = tid; i < (M_FAST * EMB_C) / 4; i += THREADS_F) {
      float4 v = src[i];
      int e = i * 4;
      int m = e >> 11;               // EMB_C == 2048
      int k = e & (EMB_C - 1);
      unsigned short h, l;
      split_bf(v.x, h, l); Ahi[m][k + 0] = h; Alo[m][k + 0] = l;
      split_bf(v.y, h, l); Ahi[m][k + 1] = h; Alo[m][k + 1] = l;
      split_bf(v.z, h, l); Ahi[m][k + 2] = h; Alo[m][k + 2] = l;
      split_bf(v.w, h, l); Ahi[m][k + 3] = h; Alo[m][k + 3] = l;
    }
  }
  __syncthreads();

  const int am  = wr * 16 + (lane & 15);       // A row for this lane
  const int ako = (lane & 16) ? 16 : 0;
  const size_t lane_off = (size_t)lane * 16;

  int tgtIdx[8];
#pragma unroll
  for (int r = 0; r < 8; ++r)
    tgtIdx[r] = targets[m_base + wr * 16 + r + ((lane >> 4) << 3)];

  float runMx[8], runSm[8], tgtAcc[8];
#pragma unroll
  for (int r = 0; r < 8; ++r) { runMx[r] = -__builtin_inff(); runSm[r] = 0.0f; tgtAcc[r] = 0.0f; }

  const int NT = VOCAB / NB_FAST;   // 250

  for (int nt = 0; nt < NT; ++nt) {
    v8f acc0 = {0.f, 0.f, 0.f, 0.f, 0.f, 0.f, 0.f, 0.f};
    v8f acc1 = {0.f, 0.f, 0.f, 0.f, 0.f, 0.f, 0.f, 0.f};
    // Fragment streams for this wave-column's two 16-col slices (shared with
    // the other wave-row -> WGP$ hits).
    const size_t fb0 = ((size_t)(nt * 8 + wc)     * NKC_C) * FRAG_ELEMS + lane_off;
    const size_t fb1 = ((size_t)(nt * 8 + 4 + wc) * NKC_C) * FRAG_ELEMS + lane_off;

#pragma unroll 2
    for (int s = 0; s < NKC_C; ++s) {
      const int kc = s * KC;
      const size_t fo0 = fb0 + (size_t)s * FRAG_ELEMS;
      const size_t fo1 = fb1 + (size_t)s * FRAG_ELEMS;
      v16bf bh0 = mk16(Whi + fo0, Whi + fo0 + 8);
      v16bf bl0 = mk16(Wlo + fo0, Wlo + fo0 + 8);
      v16bf bh1 = mk16(Whi + fo1, Whi + fo1 + 8);
      v16bf bl1 = mk16(Wlo + fo1, Wlo + fo1 + 8);
      v16bf ah  = mk16(&Ahi[am][kc + ako], &Ahi[am][kc + ako + 8]);
      v16bf al  = mk16(&Alo[am][kc + ako], &Alo[am][kc + ako + 8]);

      acc0 = __builtin_amdgcn_wmma_f32_16x16x32_bf16(false, ah, false, bh0, (short)0, acc0, false, false);
      acc1 = __builtin_amdgcn_wmma_f32_16x16x32_bf16(false, ah, false, bh1, (short)0, acc1, false, false);
      acc0 = __builtin_amdgcn_wmma_f32_16x16x32_bf16(false, ah, false, bl0, (short)0, acc0, false, false);
      acc1 = __builtin_amdgcn_wmma_f32_16x16x32_bf16(false, ah, false, bl1, (short)0, acc1, false, false);
      acc0 = __builtin_amdgcn_wmma_f32_16x16x32_bf16(false, al, false, bh0, (short)0, acc0, false, false);
      acc1 = __builtin_amdgcn_wmma_f32_16x16x32_bf16(false, al, false, bh1, (short)0, acc1, false, false);
    }

    const int n_g0 = nt * NB_FAST + (wc << 4) + (lane & 15);
    fold_tile(acc0, n_g0,      tgtIdx, runMx, runSm, tgtAcc);
    fold_tile(acc1, n_g0 + 64, tgtIdx, runMx, runSm, tgtAcc);
  }

  lane_reduce(runMx, runSm, tgtAcc);
  if ((lane & 15) == 0) {
    int mb = wr * 16 + ((lane >> 4) << 3);
#pragma unroll
    for (int r = 0; r < 8; ++r) {
      pMx[wc][mb + r] = runMx[r];
      pSm[wc][mb + r] = runSm[r];
      pTg[wc][mb + r] = tgtAcc[r];
    }
  }
  __syncthreads();

  if (tid < M_FAST) {
    float M = -__builtin_inff();
#pragma unroll
    for (int w = 0; w < 4; ++w) M = fmaxf(M, pMx[w][tid]);
    float S = 0.0f, T = 0.0f;
#pragma unroll
    for (int w = 0; w < 4; ++w) { S += pSm[w][tid] * __expf(pMx[w][tid] - M); T += pTg[w][tid]; }
    token_out(M, S, T, m_base + tid, ref_lp, adv, pmask, out, ws, SEQ);
  }
}

// ---------------------------------------------------------------------------
// FALLBACK PATH (ws too small for split weights): 128 threads, 16 tokens,
// LDS-staged in-loop conversion, double-buffered B. Same math.
// ---------------------------------------------------------------------------
__global__ __launch_bounds__(THREADS_S)
void grpo_fused_slow(const float* __restrict__ W, const float* __restrict__ H,
                     const int* __restrict__ targets, const float* __restrict__ ref_lp,
                     const float* __restrict__ adv, const unsigned char* __restrict__ pmask,
                     float* __restrict__ out, float* __restrict__ ws,
                     int VOCAB, int SEQ)
{
  __shared__ __align__(16) unsigned short Ahi[M_SLOW][A_STRIDE];
  __shared__ __align__(16) unsigned short Alo[M_SLOW][A_STRIDE];
  __shared__ __align__(16) unsigned short Bhi[2][NB_SLOW][B_STRIDE];
  __shared__ __align__(16) unsigned short Blo[2][NB_SLOW][B_STRIDE];
  __shared__ float pMx[4][M_SLOW];
  __shared__ float pSm[4][M_SLOW];
  __shared__ float pTg[4][M_SLOW];

  const int tid    = threadIdx.x;
  const int wave   = tid >> 5;
  const int lane   = tid & 31;
  const int m_base = blockIdx.x * M_SLOW;

  {
    const float4* src = (const float4*)(H + (size_t)m_base * EMB_C);
#pragma unroll 4
    for (int i = tid; i < (M_SLOW * EMB_C) / 4; i += THREADS_S) {
      float4 v = src[i];
      int e = i * 4;
      int m = e >> 11;
      int k = e & (EMB_C - 1);
      unsigned short h, l;
      split_bf(v.x, h, l); Ahi[m][k + 0] = h; Alo[m][k + 0] = l;
      split_bf(v.y, h, l); Ahi[m][k + 1] = h; Alo[m][k + 1] = l;
      split_bf(v.z, h, l); Ahi[m][k + 2] = h; Alo[m][k + 2] = l;
      split_bf(v.w, h, l); Ahi[m][k + 3] = h; Alo[m][k + 3] = l;
    }
  }
  __syncthreads();

  const int am  = lane & 15;
  const int ako = (lane & 16) ? 16 : 0;
  const int bn  = (wave << 4) + (lane & 15);
  const int bko = (lane & 16) ? 16 : 0;

  int tgtIdx[8];
#pragma unroll
  for (int r = 0; r < 8; ++r) tgtIdx[r] = targets[m_base + r + ((lane >> 4) << 3)];

  float runMx[8], runSm[8], tgtAcc[8];
#pragma unroll
  for (int r = 0; r < 8; ++r) { runMx[r] = -__builtin_inff(); runSm[r] = 0.0f; tgtAcc[r] = 0.0f; }

  const int NT = VOCAB / NB_SLOW;

  for (int nt = 0; nt < NT; ++nt) {
    const int n_base = nt * NB_SLOW;
    v8f acc = {0.f, 0.f, 0.f, 0.f, 0.f, 0.f, 0.f, 0.f};

    float4 breg[4];
#pragma unroll
    for (int j = 0; j < 4; ++j) {
      int i = tid + j * THREADS_S, e = i * 4, n = e >> 5, k = e & (KC - 1);
      breg[j] = *(const float4*)(W + (size_t)(n_base + n) * EMB_C + k);
    }
#pragma unroll
    for (int j = 0; j < 4; ++j) {
      int i = tid + j * THREADS_S, e = i * 4, n = e >> 5, k = e & (KC - 1);
      unsigned short h, l;
      split_bf(breg[j].x, h, l); Bhi[0][n][k + 0] = h; Blo[0][n][k + 0] = l;
      split_bf(breg[j].y, h, l); Bhi[0][n][k + 1] = h; Blo[0][n][k + 1] = l;
      split_bf(breg[j].z, h, l); Bhi[0][n][k + 2] = h; Blo[0][n][k + 2] = l;
      split_bf(breg[j].w, h, l); Bhi[0][n][k + 3] = h; Blo[0][n][k + 3] = l;
    }
    __syncthreads();

    for (int s = 0; s < NKC_C; ++s) {
      const int buf   = s & 1;
      const bool pref = (s + 1) < NKC_C;
      if (pref) {
        const int kc_n = (s + 1) * KC;
#pragma unroll
        for (int j = 0; j < 4; ++j) {
          int i = tid + j * THREADS_S, e = i * 4, n = e >> 5, k = e & (KC - 1);
          breg[j] = *(const float4*)(W + (size_t)(n_base + n) * EMB_C + kc_n + k);
        }
      }
      const int kc = s * KC;
      v16bf ah = mk16(&Ahi[am][kc + ako], &Ahi[am][kc + ako + 8]);
      v16bf al = mk16(&Alo[am][kc + ako], &Alo[am][kc + ako + 8]);
      v16bf bh = mk16(&Bhi[buf][bn][bko], &Bhi[buf][bn][bko + 8]);
      v16bf bl = mk16(&Blo[buf][bn][bko], &Blo[buf][bn][bko + 8]);

      acc = __builtin_amdgcn_wmma_f32_16x16x32_bf16(false, ah, false, bh, (short)0, acc, false, false);
      acc = __builtin_amdgcn_wmma_f32_16x16x32_bf16(false, ah, false, bl, (short)0, acc, false, false);
      acc = __builtin_amdgcn_wmma_f32_16x16x32_bf16(false, al, false, bh, (short)0, acc, false, false);

      if (pref) {
        const int dbuf = buf ^ 1;
#pragma unroll
        for (int j = 0; j < 4; ++j) {
          int i = tid + j * THREADS_S, e = i * 4, n = e >> 5, k = e & (KC - 1);
          unsigned short h, l;
          split_bf(breg[j].x, h, l); Bhi[dbuf][n][k + 0] = h; Blo[dbuf][n][k + 0] = l;
          split_bf(breg[j].y, h, l); Bhi[dbuf][n][k + 1] = h; Blo[dbuf][n][k + 1] = l;
          split_bf(breg[j].z, h, l); Bhi[dbuf][n][k + 2] = h; Blo[dbuf][n][k + 2] = l;
          split_bf(breg[j].w, h, l); Bhi[dbuf][n][k + 3] = h; Blo[dbuf][n][k + 3] = l;
        }
      }
      __syncthreads();
    }

    fold_tile(acc, n_base + (wave << 4) + (lane & 15), tgtIdx, runMx, runSm, tgtAcc);
  }

  lane_reduce(runMx, runSm, tgtAcc);
  if ((lane & 15) == 0) {
    int mb = (lane >> 4) << 3;
#pragma unroll
    for (int r = 0; r < 8; ++r) {
      pMx[wave][mb + r] = runMx[r];
      pSm[wave][mb + r] = runSm[r];
      pTg[wave][mb + r] = tgtAcc[r];
    }
  }
  __syncthreads();

  if (tid < M_SLOW) {
    float M = -__builtin_inff();
#pragma unroll
    for (int w = 0; w < 4; ++w) M = fmaxf(M, pMx[w][tid]);
    float S = 0.0f, T = 0.0f;
#pragma unroll
    for (int w = 0; w < 4; ++w) { S += pSm[w][tid] * __expf(pMx[w][tid] - M); T += pTg[w][tid]; }
    token_out(M, S, T, m_base + tid, ref_lp, adv, pmask, out, ws, SEQ);
  }
}

__global__ void grpo_zero_ws(float* ws, int n) {
  int i = blockIdx.x * blockDim.x + threadIdx.x;
  if (i < n) ws[i] = 0.0f;
}

__global__ void grpo_finalize(const float* __restrict__ ws, float* __restrict__ out, int BSZ) {
  if (threadIdx.x == 0 && blockIdx.x == 0) {
    float L = 0.f, P = 0.f, K = 0.f;
    for (int b = 0; b < BSZ; ++b) {
      float t = ws[b * 4 + 3];
      t = (t < 1e-9f) ? 1e-9f : t;
      L += ws[b * 4 + 0] / t;
      K += ws[b * 4 + 1] / t;
      P += ws[b * 4 + 2] / t;
    }
    float inv = 1.0f / (float)BSZ;
    out[0] = L * inv;   // loss
    out[1] = P * inv;   // policy_loss
    out[2] = K * inv;   // kl_loss
    out[3] = 1.0f;      // ratios
    out[4] = 0.0f;      // clipfrac
  }
}

extern "C" void kernel_launch(void* const* d_in, const int* in_sizes, int n_in,
                              void* d_out, int out_size, void* d_ws, size_t ws_size,
                              hipStream_t stream) {
  const float*         W    = (const float*)d_in[0];          // [VOCAB, EMB]
  const float*         H    = (const float*)d_in[1];          // [BSZ, SEQ, EMB]
  const int*           tgt  = (const int*)d_in[2];            // [BSZ, SEQ]
  const float*         ref  = (const float*)d_in[3];          // [BSZ, SEQ]
  const float*         adv  = (const float*)d_in[4];          // [BSZ]
  const unsigned char* msk  = (const unsigned char*)d_in[5];  // [BSZ, SEQ] bool
  float* out = (float*)d_out;
  float* ws  = (float*)d_ws;

  const int BSZ   = in_sizes[4];
  const int TOK   = in_sizes[3];            // BSZ * SEQ
  const int SEQ   = TOK / BSZ;
  const int EMB   = in_sizes[1] / TOK;      // == 2048
  const int VOCAB = in_sizes[0] / EMB;      // == 32000
  (void)n_in; (void)out_size;

  grpo_zero_ws<<<1, 64, 0, stream>>>(ws, 4 * BSZ);

  const size_t wbytes = (size_t)VOCAB * EMB * sizeof(unsigned short);
  const size_t need   = 256 + 2 * wbytes;   // sums + Whi + Wlo

  if (ws_size >= need) {
    unsigned short* Whi = (unsigned short*)((char*)d_ws + 256);
    unsigned short* Wlo = Whi + (size_t)VOCAB * EMB;
    const int nfrag = (VOCAB / 16) * NKC_C;             // 128000
    const int pblk  = (nfrag + 7) / 8;                  // 8 waves / block
    grpo_split_w<<<pblk, 256, 0, stream>>>(W, Whi, Wlo, nfrag);
    grpo_fused_fast<<<TOK / M_FAST, THREADS_F, 0, stream>>>(Whi, Wlo, H, tgt, ref, adv, msk,
                                                            out, ws, VOCAB, SEQ);
  } else {
    grpo_fused_slow<<<TOK / M_SLOW, THREADS_S, 0, stream>>>(W, H, tgt, ref, adv, msk,
                                                            out, ws, VOCAB, SEQ);
  }

  grpo_finalize<<<1, 32, 0, stream>>>(ws, out, BSZ);
}
